// StructureUpdate_74569222193542
// MI455X (gfx1250) — matrix operand order repaired and checked by reading, not compiled, not run on hardware
//
#include <hip/hip_runtime.h>
#include <hip/hip_bf16.h>

// ---------------------------------------------------------------------------
// IPA structure update for MI455X (gfx1250, wave32, WMMA bf16 16x16x32).
// Memory-bound on the 302MB fp32 edge tensor; we build a 151MB bf16
// transposed copy that fits MI455X's 192MB L2 so the attn@edge GEMM is
// L2-resident. Streaming fp32 edge reads use non-temporal loads so they do
// not evict the bf16 copy. All dense math uses v_wmma_f32_16x16x32_bf16.
// ---------------------------------------------------------------------------

typedef __bf16 v16bf __attribute__((ext_vector_type(16)));
typedef float  v8f   __attribute__((ext_vector_type(8)));
typedef float  v4f   __attribute__((ext_vector_type(4)));   // native vector for NT loads

// Native f32->bf16 conversion (hardware cvt on gfx1250, RNE).
__device__ inline __bf16 f2bf(float f) { return (__bf16)f; }
__device__ inline __bf16 bf_zero() {
    return __builtin_bit_cast(__bf16, (unsigned short)0);
}
__device__ inline __bf16 bflo(unsigned w) {
    return __builtin_bit_cast(__bf16, (unsigned short)(w & 0xFFFFu));
}
__device__ inline __bf16 bfhi(unsigned w) {
    return __builtin_bit_cast(__bf16, (unsigned short)(w >> 16));
}
__device__ inline __bf16 to_bf(float f)  { return f2bf(f); }
__device__ inline __bf16 to_bf(__bf16 f) { return f; }

// 8 contiguous fp32 -> fragment slots [base..base+7]
__device__ inline void ld8f(v16bf& v, int base, const float* p) {
    float4 x = ((const float4*)p)[0];
    float4 y = ((const float4*)p)[1];
    v[base+0] = f2bf(x.x); v[base+1] = f2bf(x.y);
    v[base+2] = f2bf(x.z); v[base+3] = f2bf(x.w);
    v[base+4] = f2bf(y.x); v[base+5] = f2bf(y.y);
    v[base+6] = f2bf(y.z); v[base+7] = f2bf(y.w);
}
// Non-temporal variant for streaming sources (edge fp32): TH=NT keeps L2 clean.
__device__ inline void ld8f_nt(v16bf& v, int base, const float* p) {
    v4f x = __builtin_nontemporal_load((const v4f*)p);
    v4f y = __builtin_nontemporal_load(((const v4f*)p) + 1);
    v[base+0] = f2bf(x[0]); v[base+1] = f2bf(x[1]);
    v[base+2] = f2bf(x[2]); v[base+3] = f2bf(x[3]);
    v[base+4] = f2bf(y[0]); v[base+5] = f2bf(y[1]);
    v[base+6] = f2bf(y[2]); v[base+7] = f2bf(y[3]);
}
// 16 contiguous elements -> full fragment (B-matrix path)
__device__ inline void ld16(v16bf& v, const float* p) {
    ld8f(v, 0, p); ld8f(v, 8, p + 8);
}
__device__ inline void ld16(v16bf& v, const __bf16* p) {
    uint4 u0 = ((const uint4*)p)[0];
    uint4 u1 = ((const uint4*)p)[1];
    unsigned w;
    w = u0.x; v[0]  = bflo(w); v[1]  = bfhi(w);
    w = u0.y; v[2]  = bflo(w); v[3]  = bfhi(w);
    w = u0.z; v[4]  = bflo(w); v[5]  = bfhi(w);
    w = u0.w; v[6]  = bflo(w); v[7]  = bfhi(w);
    w = u1.x; v[8]  = bflo(w); v[9]  = bfhi(w);
    w = u1.y; v[10] = bflo(w); v[11] = bfhi(w);
    w = u1.z; v[12] = bflo(w); v[13] = bfhi(w);
    w = u1.w; v[14] = bflo(w); v[15] = bfhi(w);
}

// ---------------------------------------------------------------------------
// Generic WMMA GEMM:  C[r,c] = sum_k A[r,k] * Bt[c,k]  (+bias, +residual, relu)
// A fp32 row-major [M x K] (lda), Bt [Nc x K] (ldbt) fp32 or bf16.
// C addressed as C[r*ldcr + c*ldcc]; one wave per 16x16 tile.
// ntA=1 -> A loaded with non-temporal hint (streaming, do not cache in L2).
// ---------------------------------------------------------------------------
template <typename TB>
__global__ __launch_bounds__(32)
void gemm_wmma(const float* __restrict__ A, const TB* __restrict__ Bt,
               float* __restrict__ C,
               int M, int Nc, int K, int lda, int ldbt,
               long long ldcr, long long ldcc,
               long long aBatch, long long btBatch, long long cBatch,
               const float* __restrict__ biasv, const float* __restrict__ resv,
               int relu, int ntA)
{
    long long batch = blockIdx.z;
    A  += batch * aBatch;
    Bt += batch * btBatch;
    C  += batch * cBatch;

    int m0 = blockIdx.x * 16;
    int n0 = blockIdx.y * 16;
    int lane = threadIdx.x;
    int lrow = lane & 15;
    int grp  = lane >> 4;

    int r  = m0 + lrow;
    int cN = n0 + lrow;
    bool rvalid = (r  < M);
    bool cvalid = (cN < Nc);
    const float* arow = A  + (long long)r  * lda;
    const TB*    brow = Bt + (long long)cN * ldbt;

    v8f acc = {0.f, 0.f, 0.f, 0.f, 0.f, 0.f, 0.f, 0.f};

    for (int k0 = 0; k0 < K; k0 += 32) {
        v16bf a, b;
        // A fragment: row lrow, 16 contiguous K at k0+grp*16
        if (rvalid && (k0 + 32) <= K) {
            if (ntA) {
                ld8f_nt(a, 0, arow + k0 + grp * 16);
                ld8f_nt(a, 8, arow + k0 + grp * 16 + 8);
            } else {
                ld8f(a, 0, arow + k0 + grp * 16);
                ld8f(a, 8, arow + k0 + grp * 16 + 8);
            }
        } else {
            #pragma unroll
            for (int i = 0; i < 16; ++i) {
                int kk = k0 + grp * 16 + i;
                a[i] = (rvalid && kk < K) ? f2bf(arow[kk]) : bf_zero();
            }
        }
        // B fragment: column lrow, 16 contiguous K at k0+grp*16
        if (cvalid && (k0 + 32) <= K) {
            ld16(b, brow + k0 + grp * 16);
        } else {
            #pragma unroll
            for (int i = 0; i < 16; ++i) {
                int kk = k0 + grp * 16 + i;
                b[i] = (cvalid && kk < K) ? to_bf(brow[kk]) : bf_zero();
            }
        }
        acc = __builtin_amdgcn_wmma_f32_16x16x32_bf16(
            false, a, false, b, (short)0, acc, false, false);
    }

    if (cvalid) {
        #pragma unroll
        for (int j = 0; j < 8; ++j) {
            int rr = m0 + grp * 8 + j;
            if (rr < M) {
                float v = acc[j];
                if (biasv) v += biasv[cN];
                if (resv)  v += resv[(long long)rr * Nc + cN];
                if (relu)  v = fmaxf(v, 0.f);
                C[(long long)rr * ldcr + (long long)cN * ldcc] = v;
            }
        }
    }
}

// ---------------------------------------------------------------------------
// fp32 transpose: W[R][C] -> WT[C][R]
// ---------------------------------------------------------------------------
__global__ void transpose_f32(const float* __restrict__ W, float* __restrict__ WT,
                              int R, int C)
{
    __shared__ float tile[32][33];
    int c0 = blockIdx.x * 32, r0 = blockIdx.y * 32;
    int tx = threadIdx.x & 31, ty = threadIdx.x >> 5;
    for (int yy = ty; yy < 32; yy += 8) {
        int r = r0 + yy, c = c0 + tx;
        tile[yy][tx] = (r < R && c < C) ? W[(long long)r * C + c] : 0.f;
    }
    __syncthreads();
    for (int yy = ty; yy < 32; yy += 8) {
        int c = c0 + yy, r = r0 + tx;
        if (c < C && r < R) WT[(long long)c * R + r] = tile[tx][yy];
    }
}

// edge[i][j][d] fp32 -> ebfT[i][d][j] bf16  (151MB, fits MI455X 192MB L2).
// fp32 source is streamed with NT loads so it does not sweep L2.
__global__ void edge_to_bf16T(const float* __restrict__ E, __bf16* __restrict__ T)
{
    int i = blockIdx.z;
    int j0 = blockIdx.x * 32, d0 = blockIdx.y * 32;
    __shared__ float tile[32][33];
    int tx = threadIdx.x & 31, ty = threadIdx.x >> 5;
    const float* Ei = E + (size_t)i * 768 * 128;
    for (int yy = ty; yy < 32; yy += 8)
        tile[yy][tx] =
            __builtin_nontemporal_load(Ei + (size_t)(j0 + yy) * 128 + d0 + tx);
    __syncthreads();
    __bf16* Ti = T + (size_t)i * 128 * 768;
    for (int yy = ty; yy < 32; yy += 8)
        Ti[(size_t)(d0 + yy) * 768 + j0 + tx] = f2bf(tile[tx][yy]);
}

// ---------------------------------------------------------------------------
// Point geometry: gk (global key points + |.|^2), gv transposed, sv transposed
// lin1 layout per row i (1152): q[0:192] k[192:384] v[384:576]
//                               vq[576:720] vk[720:864] vv[864:1152]
// ---------------------------------------------------------------------------
__global__ void points_kernel(const float* __restrict__ lin1,
                              const float* __restrict__ rot,
                              const float* __restrict__ trans,
                              float* __restrict__ gk4,   // [12][768][4] float4 {x,y,z,|.|^2}
                              float* __restrict__ gvT,   // [12][24][768]
                              float* __restrict__ svT)   // [12][16][768]
{
    int tid = blockIdx.x * blockDim.x + threadIdx.x;
    if (tid >= 12 * 768) return;
    int h = tid / 768, i = tid % 768;
    const float* L = lin1 + (size_t)i * 1152;
    float R[9];
    #pragma unroll
    for (int k = 0; k < 9; ++k) R[k] = rot[i * 9 + k];
    float t0 = trans[i * 3 + 0], t1 = trans[i * 3 + 1], t2 = trans[i * 3 + 2];

    #pragma unroll
    for (int p = 0; p < 4; ++p) {
        const float* y = L + 720 + (h * 4 + p) * 3;
        float x0 = R[0]*y[0] + R[1]*y[1] + R[2]*y[2] + t0;
        float x1 = R[3]*y[0] + R[4]*y[1] + R[5]*y[2] + t1;
        float x2 = R[6]*y[0] + R[7]*y[1] + R[8]*y[2] + t2;
        float sq = x0*x0 + x1*x1 + x2*x2;
        ((float4*)gk4)[(size_t)(h * 768 + i) * 4 + p] = make_float4(x0, x1, x2, sq);
    }
    #pragma unroll
    for (int p = 0; p < 8; ++p) {
        const float* y = L + 864 + (h * 8 + p) * 3;
        float x0 = R[0]*y[0] + R[1]*y[1] + R[2]*y[2] + t0;
        float x1 = R[3]*y[0] + R[4]*y[1] + R[5]*y[2] + t1;
        float x2 = R[6]*y[0] + R[7]*y[1] + R[8]*y[2] + t2;
        gvT[((size_t)h * 24 + p * 3 + 0) * 768 + i] = x0;
        gvT[((size_t)h * 24 + p * 3 + 1) * 768 + i] = x1;
        gvT[((size_t)h * 24 + p * 3 + 2) * 768 + i] = x2;
    }
    #pragma unroll
    for (int d = 0; d < 16; ++d)
        svT[((size_t)h * 16 + d) * 768 + i] = L[384 + h * 16 + d];
}

// ---------------------------------------------------------------------------
// Logits (scalar + edge bias - point distance) + softmax over j, one row per block.
// ---------------------------------------------------------------------------
__global__ void softmax_kernel(const float* __restrict__ nsb,   // [12][768][768] raw q.k
                               const float* __restrict__ biasb, // [12][768*768]
                               const float* __restrict__ gk4,   // [12][768][4] float4
                               const float* __restrict__ pw,    // [12]
                               float* __restrict__ attn)        // [12][768][768]
{
    const long long NN = 768LL * 768;
    int i = blockIdx.x, h = blockIdx.y;
    const float invs3 = 0.57735026918962576f;      // 3^-0.5
    const float wc    = 0.11785113019775793f;      // sqrt(2/PQ)/6
    float wch = wc * pw[h];

    const float4* gki = (const float4*)(gk4 + ((size_t)(h * 768 + i) * 16));
    float4 gi[4];
    #pragma unroll
    for (int p = 0; p < 4; ++p) gi[p] = gki[p];

    const float* nsrow = nsb   + (size_t)h * NN + (size_t)i * 768;
    const float* brow  = biasb + (size_t)h * NN + (size_t)i * 768;
    int t = threadIdx.x;

    float l[3];
    #pragma unroll
    for (int u = 0; u < 3; ++u) {
        int j = t + u * 256;
        const float4* gkj = (const float4*)(gk4 + ((size_t)(h * 768 + j) * 16));
        float dsum = 0.f;
        #pragma unroll
        for (int p = 0; p < 4; ++p) {
            float4 gj = gkj[p];
            float d2 = gi[p].w + gj.w -
                       2.f * (gi[p].x * gj.x + gi[p].y * gj.y + gi[p].z * gj.z);
            dsum += sqrtf(fmaxf(d2, 0.f));
        }
        l[u] = invs3 * (0.25f * nsrow[j] + brow[j] - wch * dsum);
    }

    __shared__ float red[256];
    float m = fmaxf(fmaxf(l[0], l[1]), l[2]);
    red[t] = m; __syncthreads();
    for (int st = 128; st > 0; st >>= 1) {
        if (t < st) red[t] = fmaxf(red[t], red[t + st]);
        __syncthreads();
    }
    m = red[0]; __syncthreads();

    float e[3], ssum = 0.f;
    #pragma unroll
    for (int u = 0; u < 3; ++u) { e[u] = __expf(l[u] - m); ssum += e[u]; }
    red[t] = ssum; __syncthreads();
    for (int st = 128; st > 0; st >>= 1) {
        if (t < st) red[t] += red[t + st];
        __syncthreads();
    }
    float invZ = 1.f / red[0];

    float* arow = attn + (size_t)h * NN + (size_t)i * 768;
    #pragma unroll
    for (int u = 0; u < 3; ++u) arow[t + u * 256] = e[u] * invZ;
}

// ---------------------------------------------------------------------------
// local = R^T (av - t), vnorm; write [vnorm,x,y,z] into comb cols 1728..2111
// ---------------------------------------------------------------------------
__global__ void vec_kernel(const float* __restrict__ avb,  // [12][768][24]
                           const float* __restrict__ rot,
                           const float* __restrict__ trans,
                           float* __restrict__ comb)       // [768][2112]
{
    int tid = blockIdx.x * blockDim.x + threadIdx.x;
    if (tid >= 12 * 768 * 8) return;
    int p = tid % 8;
    int i = (tid / 8) % 768;
    int h = tid / (8 * 768);
    const float* av = avb + ((size_t)h * 768 + i) * 24 + p * 3;
    float d0 = av[0] - trans[i * 3 + 0];
    float d1 = av[1] - trans[i * 3 + 1];
    float d2 = av[2] - trans[i * 3 + 2];
    const float* R = rot + i * 9;
    float lx = R[0] * d0 + R[3] * d1 + R[6] * d2;
    float ly = R[1] * d0 + R[4] * d1 + R[7] * d2;
    float lz = R[2] * d0 + R[5] * d1 + R[8] * d2;
    float nrm = sqrtf(lx * lx + ly * ly + lz * lz);
    float* o = comb + (size_t)i * 2112 + 1728 + (h * 8 + p) * 4;
    o[0] = nrm; o[1] = lx; o[2] = ly; o[3] = lz;
}

// ---------------------------------------------------------------------------
// LayerNorm over D per row
// ---------------------------------------------------------------------------
__global__ void ln_kernel(const float* __restrict__ X, const float* __restrict__ g,
                          const float* __restrict__ b, float* __restrict__ Y, int D)
{
    int row = blockIdx.x;
    const float* x = X + (size_t)row * D;
    __shared__ float red[128];
    int t = threadIdx.x;

    float s = 0.f;
    for (int c = t; c < D; c += 128) s += x[c];
    red[t] = s; __syncthreads();
    for (int st = 64; st > 0; st >>= 1) {
        if (t < st) red[t] += red[t + st];
        __syncthreads();
    }
    float mu = red[0] / D; __syncthreads();

    float v = 0.f;
    for (int c = t; c < D; c += 128) { float d = x[c] - mu; v += d * d; }
    red[t] = v; __syncthreads();
    for (int st = 64; st > 0; st >>= 1) {
        if (t < st) red[t] += red[t + st];
        __syncthreads();
    }
    float inv = rsqrtf(red[0] / D + 1e-5f);
    for (int c = t; c < D; c += 128)
        Y[(size_t)row * D + c] = (x[c] - mu) * inv * g[c] + b[c];
}

// ---------------------------------------------------------------------------
// Backbone update: corr = s @ w_bb + b_bb; quaternion -> Rc; compose frames.
// ---------------------------------------------------------------------------
__global__ void bb_kernel(const float* __restrict__ s, const float* __restrict__ wbb,
                          const float* __restrict__ bbb, const float* __restrict__ rot,
                          const float* __restrict__ trans,
                          float* __restrict__ outRot, float* __restrict__ outTr)
{
    int i = blockIdx.x * blockDim.x + threadIdx.x;
    if (i >= 768) return;
    const float* sr = s + (size_t)i * 384;
    float co[6];
    #pragma unroll
    for (int m = 0; m < 6; ++m) co[m] = bbb[m];
    for (int k = 0; k < 384; ++k) {
        float v = sr[k];
        const float* w = wbb + k * 6;
        #pragma unroll
        for (int m = 0; m < 6; ++m) co[m] += v * w[m];
    }
    float qn = sqrtf(1.f + co[0]*co[0] + co[1]*co[1] + co[2]*co[2]);
    float a = 1.f / qn, bq = co[0] / qn, cq = co[1] / qn, dq = co[2] / qn;
    float Rc[9] = {
        a*a + bq*bq - cq*cq - dq*dq, 2*bq*cq - 2*a*dq,            2*bq*dq + 2*a*cq,
        2*bq*cq + 2*a*dq,            a*a - bq*bq + cq*cq - dq*dq, 2*cq*dq - 2*a*bq,
        2*bq*dq - 2*a*cq,            2*cq*dq + 2*a*bq,            a*a - bq*bq - cq*cq + dq*dq };
    const float* R = rot + i * 9;
    #pragma unroll
    for (int r = 0; r < 3; ++r) {
        #pragma unroll
        for (int c = 0; c < 3; ++c)
            outRot[i * 9 + r * 3 + c] =
                R[r*3+0]*Rc[0*3+c] + R[r*3+1]*Rc[1*3+c] + R[r*3+2]*Rc[2*3+c];
        outTr[i * 3 + r] = R[r*3+0]*co[3] + R[r*3+1]*co[4] + R[r*3+2]*co[5]
                         + trans[i * 3 + r];
    }
}

// ---------------------------------------------------------------------------
// Host launcher
// ---------------------------------------------------------------------------
extern "C" void kernel_launch(void* const* d_in, const int* in_sizes, int n_in,
                              void* d_out, int out_size, void* d_ws, size_t ws_size,
                              hipStream_t stream)
{
    (void)in_sizes; (void)n_in; (void)out_size; (void)ws_size;
    const float* node    = (const float*)d_in[0];
    const float* edge    = (const float*)d_in[1];
    const float* rot     = (const float*)d_in[2];
    const float* trans   = (const float*)d_in[3];
    const float* w_qkv   = (const float*)d_in[4];
    const float* w_vqk   = (const float*)d_in[5];
    const float* w_vv    = (const float*)d_in[6];
    const float* w_eb    = (const float*)d_in[7];
    const float* pw      = (const float*)d_in[8];
    const float* w_final = (const float*)d_in[9];
    const float* b_final = (const float*)d_in[10];
    const float* ln1_g   = (const float*)d_in[11];
    const float* ln1_b   = (const float*)d_in[12];
    const float* ln2_g   = (const float*)d_in[13];
    const float* ln2_b   = (const float*)d_in[14];
    const float* w_r1    = (const float*)d_in[15];
    const float* b_r1    = (const float*)d_in[16];
    const float* w_r2    = (const float*)d_in[17];
    const float* b_r2    = (const float*)d_in[18];
    const float* w_r3    = (const float*)d_in[19];
    const float* b_r3    = (const float*)d_in[20];
    const float* w_bb    = (const float*)d_in[21];
    const float* b_bb    = (const float*)d_in[22];
    float* out = (float*)d_out;

    const long long NN = 768LL * 768;

    char* base = (char*)d_ws;
    size_t off = 0;
    auto take = [&](size_t bytes) -> void* {
        void* p = base + off;
        off += (bytes + 255) & ~(size_t)255;
        return p;
    };

    float*  WALLT = (float*)take(1152ull * 384 * 4);   // [qkvT | vqkT | vvT]
    float*  webT  = (float*)take(12ull * 128 * 4);
    float*  wfinT = (float*)take(384ull * 2112 * 4);
    float*  wr1T  = (float*)take(768ull * 384 * 4);
    float*  wr2T  = (float*)take(768ull * 768 * 4);
    float*  wr3T  = (float*)take(384ull * 768 * 4);
    float*  lin1  = (float*)take(768ull * 1152 * 4);
    __bf16* ebfT  = (__bf16*)take(768ull * 128 * 768 * 2);
    float*  biasb = (float*)take(12ull * NN * 4);
    float*  nsb   = (float*)take(12ull * NN * 4);
    float*  attn  = (float*)take(12ull * NN * 4);
    float*  gk4   = (float*)take(12ull * 768 * 16 * 4);
    float*  gvT   = (float*)take(12ull * 24 * 768 * 4);
    float*  svT   = (float*)take(12ull * 16 * 768 * 4);
    float*  comb  = (float*)take(768ull * 2112 * 4);
    float*  avb   = (float*)take(12ull * 768 * 24 * 4);
    float*  spre  = (float*)take(768ull * 384 * 4);
    float*  sbuf  = (float*)take(768ull * 384 * 4);
    float*  h1    = (float*)take(768ull * 768 * 4);
    float*  h2    = (float*)take(768ull * 768 * 4);
    float*  s2    = (float*)take(768ull * 384 * 4);

    // --- weight transposes (fp32) ---
    transpose_f32<<<dim3(18, 12), 256, 0, stream>>>(w_qkv, WALLT,             384, 576);
    transpose_f32<<<dim3(9, 12),  256, 0, stream>>>(w_vqk, WALLT + 576 * 384, 384, 288);
    transpose_f32<<<dim3(9, 12),  256, 0, stream>>>(w_vv,  WALLT + 864 * 384, 384, 288);
    transpose_f32<<<dim3(1, 4),   256, 0, stream>>>(w_eb,  webT,  128, 12);
    transpose_f32<<<dim3(12, 66), 256, 0, stream>>>(w_final, wfinT, 2112, 384);
    transpose_f32<<<dim3(24, 12), 256, 0, stream>>>(w_r1, wr1T, 384, 768);
    transpose_f32<<<dim3(24, 24), 256, 0, stream>>>(w_r2, wr2T, 768, 768);
    transpose_f32<<<dim3(12, 24), 256, 0, stream>>>(w_r3, wr3T, 768, 384);

    // --- bf16 transposed edge copy (fits in L2; NT reads of fp32 source) ---
    edge_to_bf16T<<<dim3(24, 4, 768), 256, 0, stream>>>(edge, ebfT);

    // --- lin1 = node @ [w_qkv|w_vqk|w_vv]  (768 x 1152, K=384) ---
    gemm_wmma<float><<<dim3(48, 72, 1), 32, 0, stream>>>(
        node, WALLT, lin1, 768, 1152, 384, 384, 384,
        1152, 1, 0, 0, 0, nullptr, nullptr, 0, 0);

    // --- edge bias: biasb[h][(i,j)] = edge_rows @ w_eb (589824 x 12, K=128).
    //     A is the 302MB fp32 edge stream -> non-temporal loads. ---
    gemm_wmma<float><<<dim3(36864, 1, 1), 32, 0, stream>>>(
        edge, webT, biasb, (int)NN, 12, 128, 128, 128,
        1, NN, 0, 0, 0, nullptr, nullptr, 0, 1);

    // --- point geometry ---
    points_kernel<<<36, 256, 0, stream>>>(lin1, rot, trans, gk4, gvT, svT);

    // --- node_scalar per head: q_h @ k_h^T (768x768, K=16 zero-padded to 32) ---
    gemm_wmma<float><<<dim3(48, 48, 12), 32, 0, stream>>>(
        lin1, lin1 + 192, nsb, 768, 768, 16, 1152, 1152,
        768, 1, 16, 16, NN, nullptr, nullptr, 0, 0);

    // --- logits + softmax ---
    softmax_kernel<<<dim3(768, 12), 256, 0, stream>>>(nsb, biasb, gk4, pw, attn);

    // --- edge_out: per-i attn_i[12x768] @ E_i[768x128] from bf16 transposed edge ---
    gemm_wmma<__bf16><<<dim3(1, 8, 768), 32, 0, stream>>>(
        attn, ebfT, comb, 12, 128, 768, (int)NN, 768,
        128, 1, 768, 128LL * 768, 2112, nullptr, nullptr, 0, 0);

    // --- scalar_out: per-h attn_h[768x768] @ sv_h[768x16] -> comb cols 1536.. ---
    gemm_wmma<float><<<dim3(48, 1, 12), 32, 0, stream>>>(
        attn, svT, comb + 1536, 768, 16, 768, 768, 768,
        2112, 1, NN, 16LL * 768, 16, nullptr, nullptr, 0, 0);

    // --- av: per-h attn_h @ gv_h[768x24] ---
    gemm_wmma<float><<<dim3(48, 2, 12), 32, 0, stream>>>(
        attn, gvT, avb, 768, 24, 768, 768, 768,
        24, 1, NN, 24LL * 768, 768LL * 24, nullptr, nullptr, 0, 0);

    // --- local point coords + norms into comb cols 1728.. ---
    vec_kernel<<<288, 256, 0, stream>>>(avb, rot, trans, comb);

    // --- final projection + residual ---
    gemm_wmma<float><<<dim3(48, 24, 1), 32, 0, stream>>>(
        comb, wfinT, spre, 768, 384, 2112, 2112, 2112,
        384, 1, 0, 0, 0, b_final, node, 0, 0);

    ln_kernel<<<768, 128, 0, stream>>>(spre, ln1_g, ln1_b, sbuf, 384);

    // --- transition MLP ---
    gemm_wmma<float><<<dim3(48, 48, 1), 32, 0, stream>>>(
        sbuf, wr1T, h1, 768, 768, 384, 384, 384,
        768, 1, 0, 0, 0, b_r1, nullptr, 1, 0);
    gemm_wmma<float><<<dim3(48, 48, 1), 32, 0, stream>>>(
        h1, wr2T, h2, 768, 768, 768, 768, 768,
        768, 1, 0, 0, 0, b_r2, nullptr, 1, 0);
    gemm_wmma<float><<<dim3(48, 24, 1), 32, 0, stream>>>(
        h2, wr3T, s2, 768, 384, 768, 768, 768,
        384, 1, 0, 0, 0, b_r3, sbuf, 0, 0);

    ln_kernel<<<768, 128, 0, stream>>>(s2, ln2_g, ln2_b, out, 384);

    // --- backbone update ---
    bb_kernel<<<3, 256, 0, stream>>>(out, w_bb, b_bb, rot, trans,
                                     out + 294912, out + 294912 + 6912);
}
